// Backbone_1314259992835
// MI455X (gfx1250) — matrix operand order repaired and checked by reading, not compiled, not run on hardware
//
#include <hip/hip_runtime.h>
#include <hip/hip_bf16.h>
#include <stdint.h>

// ---------------------------------------------------------------------------
// Point Transformer backbone for MI455X (gfx1250, wave32, WMMA + TDM).
// Heavy GEMMs:  v_wmma_f32_16x16x32_bf16; 32x32 bf16 A tile staged into LDS
//               by the Tensor Data Mover (double-buffered, s_wait_tensorcnt);
//               each wave computes a 32x64 tile: 2 A frags x 4 B frags =
//               8 straight-line WMMAs per K-step (no divergence), f32 accum.
// Irregular ops (KNN top-16, FPS, gathers, softmax-reduce, BN, maxpool):
//               scalar VALU kernels (bandwidth/latency bound anyway).
// ---------------------------------------------------------------------------

typedef __bf16 bf16;
typedef __attribute__((ext_vector_type(16))) __bf16 v16bf;
typedef __attribute__((ext_vector_type(8)))  __bf16 v8bf;
typedef __attribute__((ext_vector_type(8)))  float  v8f;
typedef __attribute__((ext_vector_type(4))) unsigned int u32x4;
typedef __attribute__((ext_vector_type(4))) int i32x4;
typedef __attribute__((ext_vector_type(8))) int i32x8;

#define DMODEL 512
#define KNB    16      // K_NEIGH

#ifndef __has_builtin
#define __has_builtin(x) 0
#endif
#if __has_builtin(__builtin_amdgcn_tensor_load_to_lds) && \
    __has_builtin(__builtin_amdgcn_s_wait_tensorcnt)
#define HAVE_TDM 1
#else
#define HAVE_TDM 0
#endif

#define GS(i, n) for (long i = (long)blockIdx.x * blockDim.x + threadIdx.x; \
                      i < (n); i += (long)gridDim.x * blockDim.x)

static inline dim3 gs(long n) {
  long b = (n + 255) / 256;
  if (b > 32768) b = 32768;
  if (b < 1) b = 1;
  return dim3((unsigned)b);
}

// ------------------------------ conversions --------------------------------
__global__ void k_f32_to_bf16(const float* __restrict__ in, bf16* __restrict__ out, long n) {
  GS(i, n) out[i] = (bf16)in[i];
}

// ------------------------------- TDM helper --------------------------------
#if HAVE_TDM
// Issue a TDM load of a (rows x 32) bf16 tile (row stride Kd elems) into LDS.
// D# layout per CDNA5 ISA:
//   g0: count=1 | lds_addr | global_addr[56:0] | type=2
//   g1: data_size=1(2B), tensor_dim0/1 (OOB), tile_dim0=32, tile_dim1=rows,
//       tensor_dim0_stride=Kd
__device__ __forceinline__ void tdm_load_tile(unsigned ldsAddr, const bf16* g,
                                              int rows, int Kd) {
  const unsigned long long ga = (unsigned long long)(size_t)g;
  u32x4 g0;
  g0[0] = 1u;                                     // count=1
  g0[1] = ldsAddr;                                // lds_addr (bytes)
  g0[2] = (unsigned)ga;                           // global_addr[31:0]
  g0[3] = (unsigned)((ga >> 32) & 0x01ffffffu) | (2u << 30);  // [56:32] | type=2
  i32x8 g1;
  const unsigned td0 = 1u << 20;                  // generous OOB dims
  const unsigned td1 = 1u << 20;
  g1[0] = (int)(1u << 16);                        // data_size=1 (2 bytes)
  g1[1] = (int)((td0 & 0xffffu) << 16);           // tensor_dim0[15:0]
  g1[2] = (int)((td0 >> 16) | ((td1 & 0xffffu) << 16));
  g1[3] = (int)((td1 >> 16) | (32u << 16));       // tile_dim0 = 32
  g1[4] = (int)(unsigned)rows;                    // tile_dim1 = rows, tile_dim2 = 0
  g1[5] = (int)(unsigned)Kd;                      // tensor_dim0_stride lo32
  g1[6] = 0;                                      // stride hi16 | dim1_stride lo16
  g1[7] = 0;
  const i32x4 z4 = {0, 0, 0, 0};
#if defined(__clang_major__) && __clang_major__ >= 23
  const i32x8 z8 = {0, 0, 0, 0, 0, 0, 0, 0};
  __builtin_amdgcn_tensor_load_to_lds(g0, g1, z4, z4, z8, 0);
#else
  __builtin_amdgcn_tensor_load_to_lds(g0, g1, z4, z4, 0);
#endif
}
#endif

// ------------------------- WMMA bf16 GEMM (core) ---------------------------
// C[M,Nout] = A[M,Kd](bf16) * W[Nout,Kd]^T(bf16) (+bias) (+resid) (relu?)
// Requires: Kd %32==0, Nout %16==0, M % (MT*16)==0.
// Block = 256 threads = 8 waves sharing MT m-tiles; wave owns NT n-tiles.
// NT>1 instantiations have NO bounds checks: caller guarantees full tiles,
// so the MT*NT WMMAs are straight-line and accumulate in place.
template <int MT, int NT>
__global__ __launch_bounds__(256)
void k_gemm_bf16_wmma(const bf16* __restrict__ A, const bf16* __restrict__ W,
                      const float* __restrict__ bias, const float* __restrict__ resid,
                      float* __restrict__ outF, bf16* __restrict__ outB,
                      int M, int Kd, int Nout, int relu) {
  __shared__ __attribute__((aligned(32))) bf16 lA[2][MT * 16 * 32];
  const int t     = threadIdx.x;
  const int wave  = t >> 5;
  const int lane  = t & 31;
  const int hi    = lane >> 4;
  const int ln    = lane & 15;
  const int mrow0 = blockIdx.y * (MT * 16);
  const int tile0 = (blockIdx.x * 8 + wave) * NT;
  const long arow = (long)mrow0 * Kd;

  v8f c[MT][NT];
#pragma unroll
  for (int v_ = 0; v_ < MT; ++v_)
#pragma unroll
    for (int u = 0; u < NT; ++u)
      c[v_][u] = (v8f){0.f, 0.f, 0.f, 0.f, 0.f, 0.f, 0.f, 0.f};

  int buf = 0;

#if HAVE_TDM
  if (wave == 0) {
    tdm_load_tile((unsigned)(size_t)&lA[0][0], A + arow, MT * 16, Kd);
    __builtin_amdgcn_s_wait_tensorcnt((short)0);
  }
  __syncthreads();
#else
  for (int e = t; e < MT * 16 * 8; e += 256) {   // cooperative stage, 8B chunks
    const int r = e >> 3, ch = e & 7;
    *(uint2*)&lA[0][r * 32 + ch * 4] = *(const uint2*)(A + arow + (long)r * Kd + ch * 4);
  }
  __syncthreads();
#endif

  for (int kk = 0; kk < Kd; kk += 32) {
    const bool more = (kk + 32) < Kd;
#if HAVE_TDM
    if (wave == 0 && more)             // DMA next slab while we compute this one
      tdm_load_tile((unsigned)(size_t)&lA[buf ^ 1][0], A + arow + kk + 32, MT * 16, Kd);
#else
    if (more) {
      for (int e = t; e < MT * 16 * 8; e += 256) {
        const int r = e >> 3, ch = e & 7;
        *(uint2*)&lA[buf ^ 1][r * 32 + ch * 4] =
            *(const uint2*)(A + arow + (long)r * Kd + kk + 32 + ch * 4);
      }
    }
#endif
    // A fragments (ISA 7.12.2, 16-bit A 16x32): row = v_*16 + ln, K slices by hi
    union F16 { v16bf v; v8bf h[2]; };
    F16 a[MT];
#pragma unroll
    for (int v_ = 0; v_ < MT; ++v_) {
      a[v_].h[0] = *(const v8bf*)&lA[buf][(v_ * 16 + ln) * 32 + 8 * hi];
      a[v_].h[1] = *(const v8bf*)&lA[buf][(v_ * 16 + ln) * 32 + 16 + 8 * hi];
    }
#pragma unroll
    for (int u = 0; u < NT; ++u) {
      const int nt = tile0 + u;
      if (NT == 1 && nt * 16 >= Nout) continue;   // compile-time dead for NT>1
      F16 b;
      b.h[0] = *(const v8bf*)(W + (long)(nt * 16 + ln) * Kd + kk + 8 * hi);
      b.h[1] = *(const v8bf*)(W + (long)(nt * 16 + ln) * Kd + kk + 16 + 8 * hi);
#pragma unroll
      for (int v_ = 0; v_ < MT; ++v_)
        c[v_][u] = __builtin_amdgcn_wmma_f32_16x16x32_bf16(
                       false, a[v_].v, false, b.v, (short)0, c[v_][u], false, false);
    }
#if HAVE_TDM
    if (wave == 0 && more) __builtin_amdgcn_s_wait_tensorcnt((short)0);
#endif
    __syncthreads();
    buf ^= 1;
  }

#pragma unroll
  for (int u = 0; u < NT; ++u) {
    const int nt = tile0 + u;
    if (NT == 1 && nt * 16 >= Nout) continue;
    const int n = nt * 16 + ln;
    const float bi = bias ? bias[n] : 0.0f;
#pragma unroll
    for (int v_ = 0; v_ < MT; ++v_) {
#pragma unroll
      for (int j = 0; j < 8; ++j) {    // C/D: vgpr j -> row j + 8*hi
        const long m = (long)mrow0 + v_ * 16 + j + 8 * hi;
        float v = c[v_][u][j] + bi;
        if (resid) v += resid[m * Nout + n];
        if (relu)  v = fmaxf(v, 0.0f);
        if (outF)  outF[m * Nout + n] = v;
        if (outB)  outB[m * Nout + n] = (bf16)v;
      }
    }
  }
}

static void gemm(const bf16* A, const bf16* W, const float* bias, const float* resid,
                 float* outF, bf16* outB, long M, int Kd, int Nout, int relu,
                 hipStream_t s) {
  if (Nout % 512 == 0 && M % 32 == 0) {
    // 8 waves x 4 n-tiles x 16 = 512 cols, 2 m-tiles = 32 rows per block
    dim3 g((unsigned)(Nout / 512), (unsigned)(M / 32));
    k_gemm_bf16_wmma<2, 4><<<g, 256, 0, s>>>(A, W, bias, resid, outF, outB,
                                             (int)M, Kd, Nout, relu);
  } else {
    dim3 g((unsigned)((Nout + 127) / 128), (unsigned)(M / 16));
    k_gemm_bf16_wmma<1, 1><<<g, 256, 0, s>>>(A, W, bias, resid, outF, outB,
                                             (int)M, Kd, Nout, relu);
  }
}

// ----------------------------- scalar linear -------------------------------
__global__ void k_linear_scalar(const float* __restrict__ A, const float* __restrict__ W,
                                const float* __restrict__ bias, float* __restrict__ out,
                                long M, int Kd, int Nout, int relu) {
  GS(i, M * (long)Nout) {
    const long m = i / Nout;
    const int  o = (int)(i % Nout);
    float acc = bias ? bias[o] : 0.0f;
    const float* a = A + m * Kd;
    const float* w = W + (long)o * Kd;
    for (int k = 0; k < Kd; ++k) acc += a[k] * w[k];
    if (relu) acc = fmaxf(acc, 0.0f);
    out[i] = acc;
  }
}

// ------------------------------- misc ops ----------------------------------
__global__ void k_extract_xyz(const float* __restrict__ x, float* __restrict__ xyz, long n) {
  GS(i, n) {
    xyz[i * 3 + 0] = x[i * 6 + 0];
    xyz[i * 3 + 1] = x[i * 6 + 1];
    xyz[i * 3 + 2] = x[i * 6 + 2];
  }
}

// KNN: one wave per query; per-lane insertion top-16, lane0 serial merge.
__global__ __launch_bounds__(32)
void k_knn(const float* __restrict__ q, const float* __restrict__ s,
           int* __restrict__ out, int Mq, int Ns) {
  __shared__ float sd[32 * KNB];
  __shared__ int   si[32 * KNB];
  const int  lane = threadIdx.x;
  const long bq = blockIdx.x;               // b*Mq + m
  const long b  = bq / Mq;
  const float qx = q[bq * 3], qy = q[bq * 3 + 1], qz = q[bq * 3 + 2];
  const float* S = s + b * Ns * 3;
  float bd[KNB]; int bi[KNB];
#pragma unroll
  for (int k = 0; k < KNB; ++k) { bd[k] = 3.0e38f; bi[k] = 0; }
  for (int j = lane; j < Ns; j += 32) {
    const float dx = S[j * 3] - qx, dy = S[j * 3 + 1] - qy, dz = S[j * 3 + 2] - qz;
    const float d = dx * dx + dy * dy + dz * dz;
    if (d < bd[KNB - 1]) {
      int p = KNB - 1;
      while (p > 0 && bd[p - 1] > d) { bd[p] = bd[p - 1]; bi[p] = bi[p - 1]; --p; }
      bd[p] = d; bi[p] = j;
    }
  }
  for (int k = 0; k < KNB; ++k) { sd[lane * KNB + k] = bd[k]; si[lane * KNB + k] = bi[k]; }
  __syncthreads();
  if (lane == 0) {
    for (int k = 0; k < KNB; ++k) {
      float best = 3.0e38f; int bp = 0;
      for (int e = 0; e < 32 * KNB; ++e)
        if (sd[e] < best) { best = sd[e]; bp = e; }
      out[bq * KNB + k] = si[bp];
      sd[bp] = 3.4e38f;
    }
  }
}

// Farthest point sampling: one block per batch, iterative block-argmax.
__global__ __launch_bounds__(256)
void k_fps(const float* __restrict__ xyz, float* __restrict__ dist,
           int* __restrict__ fidx, int N, int npoint) {
  __shared__ float sred[256];
  __shared__ int   sidx[256];
  __shared__ int   sFar;
  const int b = blockIdx.x;
  const float* P = xyz + (long)b * N * 3;
  float* D = dist + (long)b * N;
  for (int i = threadIdx.x; i < N; i += 256) D[i] = 1e10f;
  if (threadIdx.x == 0) sFar = 0;
  __syncthreads();
  for (int it = 0; it < npoint; ++it) {
    const int far = sFar;
    if (threadIdx.x == 0) fidx[(long)b * npoint + it] = far;
    const float cx = P[far * 3], cy = P[far * 3 + 1], cz = P[far * 3 + 2];
    float bmax = -1.0f; int bi = 0;
    for (int i = threadIdx.x; i < N; i += 256) {
      const float dx = P[i * 3] - cx, dy = P[i * 3 + 1] - cy, dz = P[i * 3 + 2] - cz;
      const float dd = fminf(D[i], dx * dx + dy * dy + dz * dz);
      D[i] = dd;
      if (dd > bmax) { bmax = dd; bi = i; }
    }
    sred[threadIdx.x] = bmax; sidx[threadIdx.x] = bi;
    __syncthreads();
    for (int st = 128; st > 0; st >>= 1) {
      if (threadIdx.x < st && sred[threadIdx.x + st] > sred[threadIdx.x]) {
        sred[threadIdx.x] = sred[threadIdx.x + st];
        sidx[threadIdx.x] = sidx[threadIdx.x + st];
      }
      __syncthreads();
    }
    if (threadIdx.x == 0) sFar = sidx[0];
    __syncthreads();
  }
}

__global__ void k_gather3(const float* __restrict__ xyz, const int* __restrict__ idx,
                          float* __restrict__ out, int Ns, int Mq, long nBM) {
  GS(i, nBM) {
    const long b = i / Mq;
    const long src = (b * Ns + idx[i]) * 3;
    out[i * 3 + 0] = xyz[src + 0];
    out[i * 3 + 1] = xyz[src + 1];
    out[i * 3 + 2] = xyz[src + 2];
  }
}

__global__ void k_pos(const float* __restrict__ xyz, const int* __restrict__ knn,
                      float* __restrict__ pos, int N, long nRows) {
  GS(r, nRows) {
    const long bm = r / KNB;
    const long b  = bm / N;
    const int  j  = knn[r];
    const float* c0 = xyz + bm * 3;
    const float* c1 = xyz + (b * N + j) * 3;
    pos[r * 3 + 0] = c0[0] - c1[0];
    pos[r * 3 + 1] = c0[1] - c1[1];
    pos[r * 3 + 2] = c0[2] - c1[2];
  }
}

// t0[b,m,k,d] = q[b,m,d] - kfull[b, knn, d] + pe[b,m,k,d]   (bf16 out)
__global__ void k_make_t0(const float* __restrict__ q, const float* __restrict__ kf,
                          const float* __restrict__ pe, const int* __restrict__ knn,
                          bf16* __restrict__ t0, int N, long total) {
  GS(i, total) {
    const int  d  = (int)(i % DMODEL);
    const long r  = i / DMODEL;
    const long bm = r / KNB;
    const long b  = bm / N;
    const int  j  = knn[r];
    t0[i] = (bf16)(q[bm * DMODEL + d] - kf[(b * N + j) * (long)DMODEL + d] + pe[i]);
  }
}

// pe := v_gathered + pe   (in place)
__global__ void k_vpe(const float* __restrict__ vf, const int* __restrict__ knn,
                      float* __restrict__ pe, int N, long total) {
  GS(i, total) {
    const int  d  = (int)(i % DMODEL);
    const long r  = i / DMODEL;
    const long bm = r / KNB;
    const long b  = bm / N;
    const int  j  = knn[r];
    pe[i] = vf[(b * N + j) * (long)DMODEL + d] + pe[i];
  }
}

// softmax over neighbors (axis=-2) then weighted sum with (v+pe)
__global__ void k_attn_reduce(const float* __restrict__ attn, const float* __restrict__ vpe,
                              float* __restrict__ out, long totalBMD) {
  const float sc = 0.044194173824159216f;   // 1/sqrt(512)
  GS(i, totalBMD) {
    const long bm = i / DMODEL;
    const int  d  = (int)(i % DMODEL);
    const float* ap = attn + bm * KNB * (long)DMODEL + d;
    const float* vp = vpe  + bm * KNB * (long)DMODEL + d;
    float mx = -3.0e38f;
#pragma unroll
    for (int k = 0; k < KNB; ++k) mx = fmaxf(mx, ap[k * DMODEL] * sc);
    float s = 0.0f, acc = 0.0f;
#pragma unroll
    for (int k = 0; k < KNB; ++k) {
      const float e = __expf(ap[k * DMODEL] * sc - mx);
      s += e;
      acc += e * vp[k * DMODEL];
    }
    out[i] = acc / s;
  }
}

// h[b,m,k,:] = [xyz[knn]-newxyz , points[knn]]
__global__ void k_group_concat(const float* __restrict__ xyz, const float* __restrict__ nxyz,
                               const float* __restrict__ pts, const int* __restrict__ knn,
                               float* __restrict__ h, int Ns, int Mq, int Cin, long nRows) {
  GS(r, nRows) {
    const long bm = r / KNB;
    const long b  = bm / Mq;
    const int  j  = knn[r];
    float* hr = h + r * (long)(3 + Cin);
    const float* src = xyz + (b * Ns + j) * 3;
    const float* ctr = nxyz + bm * 3;
    hr[0] = src[0] - ctr[0];
    hr[1] = src[1] - ctr[1];
    hr[2] = src[2] - ctr[2];
    const float* pp = pts + (b * Ns + j) * (long)Cin;
    for (int c = 0; c < Cin; ++c) hr[3 + c] = pp[c];
  }
}

__global__ void k_zero(float* __restrict__ p, long n) { GS(i, n) p[i] = 0.0f; }

__global__ void k_bn_accum(const float* __restrict__ h, float* __restrict__ sums,
                           float* __restrict__ sq, long total, int C) {
  GS(i, total) {
    const int c = (int)(i % C);
    const float v = h[i];
    atomicAdd(&sums[c], v);
    atomicAdd(&sq[c], v * v);
  }
}

__global__ void k_bn_apply(float* __restrict__ h, const float* __restrict__ sums,
                           const float* __restrict__ sq, const float* __restrict__ g,
                           const float* __restrict__ beta, float cnt, long total, int C) {
  GS(i, total) {
    const int c = (int)(i % C);
    const float mean = sums[c] / cnt;
    const float var  = sq[c] / cnt - mean * mean;
    const float v = (h[i] - mean) * rsqrtf(var + 1e-5f) * g[c] + beta[c];
    h[i] = fmaxf(v, 0.0f);
  }
}

__global__ void k_maxpool(const float* __restrict__ h, float* __restrict__ out,
                          int C, long nBM) {
  GS(i, nBM * (long)C) {
    const long bm = i / C;
    const int  c  = (int)(i % C);
    float m = -3.0e38f;
    const float* hp = h + bm * KNB * (long)C + c;
#pragma unroll
    for (int k = 0; k < KNB; ++k) m = fmaxf(m, hp[k * C]);
    out[i] = m;
  }
}

// ------------------------------ host side ----------------------------------
struct Bump {
  char* base; size_t off, cap;
  void* get(size_t n) { size_t a = (off + 255) & ~(size_t)255; off = a + n; return base + a; }
  float* f(long n) { return (float*)get((size_t)n * 4); }
  bf16*  h(long n) { return (bf16*) get((size_t)n * 2); }
  int*   i(long n) { return (int*)  get((size_t)n * 4); }
};

static bf16* tobf(Bump& w, const float* src, long n, hipStream_t s) {
  bf16* d = w.h(n);
  k_f32_to_bf16<<<gs(n), 256, 0, s>>>(src, d, n);
  return d;
}

struct Lin { const float* W; const float* b; };
struct TBp { Lin d1, d2, f1, f2, g1, g2; const float* wk; const float* wq; const float* wv; };

// Bump passed BY VALUE: all scratch inside is released on return.
static void run_tb(const TBp& P, const float* xyz, const float* fin, float* fout,
                   int B, int N, int dp, Bump w, hipStream_t s) {
  const long BN = (long)B * N;
  const long R  = BN * KNB;

  int* knn = w.i(R);
  k_knn<<<dim3((unsigned)BN), 32, 0, s>>>(xyz, xyz, knn, N, N);

  bf16* f1W = tobf(w, P.f1.W, (long)DMODEL * dp, s);
  bf16* wqW = tobf(w, P.wq, (long)DMODEL * DMODEL, s);
  bf16* wkW = tobf(w, P.wk, (long)DMODEL * DMODEL, s);
  bf16* wvW = tobf(w, P.wv, (long)DMODEL * DMODEL, s);
  bf16* d2W = tobf(w, P.d2.W, (long)DMODEL * DMODEL, s);
  bf16* g1W = tobf(w, P.g1.W, (long)DMODEL * DMODEL, s);
  bf16* g2W = tobf(w, P.g2.W, (long)DMODEL * DMODEL, s);
  bf16* f2W = tobf(w, P.f2.W, (long)dp * DMODEL, s);

  bf16*  finB = tobf(w, fin, BN * dp, s);
  float* x    = w.f(BN * DMODEL);
  bf16*  xB   = w.h(BN * DMODEL);
  gemm(finB, f1W, P.f1.b, nullptr, x, xB, BN, dp, DMODEL, 0, s);

  float* q  = w.f(BN * DMODEL);
  float* kf = w.f(BN * DMODEL);
  float* vf = w.f(BN * DMODEL);
  gemm(xB, wqW, nullptr, nullptr, q,  nullptr, BN, DMODEL, DMODEL, 0, s);
  gemm(xB, wkW, nullptr, nullptr, kf, nullptr, BN, DMODEL, DMODEL, 0, s);
  gemm(xB, wvW, nullptr, nullptr, vf, nullptr, BN, DMODEL, DMODEL, 0, s);

  float* pos = w.f(R * 3);
  k_pos<<<gs(R), 256, 0, s>>>(xyz, knn, pos, N, R);

  float* bigA = w.f(R * DMODEL);          // delta1-out, later attn
  bf16*  bigC = w.h(R * DMODEL);          // delta1-out bf16, later t0
  k_linear_scalar<<<gs(R * DMODEL), 256, 0, s>>>(pos, P.d1.W, P.d1.b, bigA, R, 3, DMODEL, 1);
  k_f32_to_bf16<<<gs(R * DMODEL), 256, 0, s>>>(bigA, bigC, R * DMODEL);

  float* pe = w.f(R * DMODEL);
  gemm(bigC, d2W, P.d2.b, nullptr, pe, nullptr, R, DMODEL, DMODEL, 0, s);

  k_make_t0<<<gs(R * DMODEL), 256, 0, s>>>(q, kf, pe, knn, bigC, N, R * DMODEL);  // bigC := t0
  bf16* h1 = w.h(R * DMODEL);
  gemm(bigC, g1W, P.g1.b, nullptr, nullptr, h1, R, DMODEL, DMODEL, 1, s);          // relu, bf16 only
  gemm(h1, g2W, P.g2.b, nullptr, bigA, nullptr, R, DMODEL, DMODEL, 0, s);          // bigA := attn
  k_vpe<<<gs(R * DMODEL), 256, 0, s>>>(vf, knn, pe, N, R * DMODEL);                // pe := v+pe

  float* res = w.f(BN * DMODEL);
  k_attn_reduce<<<gs(BN * DMODEL), 256, 0, s>>>(bigA, pe, res, BN * DMODEL);
  bf16* resB = tobf(w, res, BN * DMODEL, s);
  gemm(resB, f2W, P.f2.b, fin, fout, nullptr, BN, DMODEL, dp, 0, s);               // +residual
}

struct SAl { const float* W; const float* b; const float* g; const float* be; int ci, co; };

static void run_sa(const SAl L[2], const float* xyz, const float* pts,
                   int B, int Ns, int Cin, int Mq,
                   float* nxyz, float* npts, Bump w, hipStream_t s) {
  const long BM = (long)B * Mq;
  float* dist = w.f((long)B * Ns);
  int*   fidx = w.i(BM);
  k_fps<<<dim3((unsigned)B), 256, 0, s>>>(xyz, dist, fidx, Ns, Mq);
  k_gather3<<<gs(BM), 256, 0, s>>>(xyz, fidx, nxyz, Ns, Mq, BM);

  int* knn = w.i(BM * KNB);
  k_knn<<<dim3((unsigned)BM), 32, 0, s>>>(nxyz, xyz, knn, Mq, Ns);

  const long R = BM * KNB;
  const int  C0 = 3 + Cin;
  float* h0 = w.f(R * C0);
  k_group_concat<<<gs(R), 256, 0, s>>>(xyz, nxyz, pts, knn, h0, Ns, Mq, Cin, R);

  float* cur = h0;
  int Cc = C0;
  for (int l = 0; l < 2; ++l) {
    const int Co = L[l].co;
    float* nx = w.f(R * Co);
    k_linear_scalar<<<gs(R * Co), 256, 0, s>>>(cur, L[l].W, L[l].b, nx, R, Cc, Co, 0);
    float* stats = w.f(2 * Co);
    k_zero<<<gs(2 * Co), 256, 0, s>>>(stats, 2 * Co);
    k_bn_accum<<<gs(R * Co), 256, 0, s>>>(nx, stats, stats + Co, R * Co, Co);
    k_bn_apply<<<gs(R * Co), 256, 0, s>>>(nx, stats, stats + Co, L[l].g, L[l].be,
                                          (float)R, R * Co, Co);
    cur = nx; Cc = Co;
  }
  k_maxpool<<<gs(BM * Cc), 256, 0, s>>>(cur, npts, Cc, BM);
}

extern "C" void kernel_launch(void* const* d_in, const int* in_sizes, int n_in,
                              void* d_out, int out_size, void* d_ws, size_t ws_size,
                              hipStream_t stream) {
  (void)in_sizes; (void)n_in; (void)out_size;
  // Input order assumes jax pytree flattening (dict keys sorted) of params
  // after x:  fc1a, fc1b, sa1[0..1]{W,b,beta,g}, sa2[..], tr{delta1,delta2,
  // fc1,fc2,gamma1,gamma2,wk,wq,wv} x3.
  auto F = [&](int i) { return (const float*)d_in[i]; };
  const float* x = F(0);

  auto mkTB = [&](int t) {
    TBp p;
    p.d1 = {F(t + 0),  F(t + 1)};
    p.d2 = {F(t + 2),  F(t + 3)};
    p.f1 = {F(t + 4),  F(t + 5)};
    p.f2 = {F(t + 6),  F(t + 7)};
    p.g1 = {F(t + 8),  F(t + 9)};
    p.g2 = {F(t + 10), F(t + 11)};
    p.wk = F(t + 12); p.wq = F(t + 13); p.wv = F(t + 14);
    return p;
  };

  Bump ws{(char*)d_ws, 0, ws_size};
  const int B = 2, N0 = 4096, N1 = 1024, N2 = 256;

  // persistent buffers
  float* xyz0  = ws.f((long)B * N0 * 3);
  float* xyz1  = ws.f((long)B * N1 * 3);
  float* xyz2  = ws.f((long)B * N2 * 3);
  float* tmp   = ws.f((long)B * N0 * 32);
  float* pts0  = ws.f((long)B * N0 * 32);
  float* pts0o = ws.f((long)B * N0 * 32);
  float* pts1  = ws.f((long)B * N1 * 64);
  float* pts1o = ws.f((long)B * N1 * 64);
  float* pts2  = ws.f((long)B * N2 * 128);

  k_extract_xyz<<<gs((long)B * N0), 256, 0, stream>>>(x, xyz0, (long)B * N0);
  // feats = fc1b( relu( fc1a(x) ) )
  k_linear_scalar<<<gs((long)B * N0 * 32), 256, 0, stream>>>(x,   F(1), F(2), tmp,  (long)B * N0, 6,  32, 1);
  k_linear_scalar<<<gs((long)B * N0 * 32), 256, 0, stream>>>(tmp, F(3), F(4), pts0, (long)B * N0, 32, 32, 0);

  TBp tr0 = mkTB(21);
  run_tb(tr0, xyz0, pts0, pts0o, B, N0, 32, ws, stream);

  SAl sa1[2] = {{F(5), F(6),  F(8),  F(7),  35, 64},
                {F(9), F(10), F(12), F(11), 64, 64}};
  run_sa(sa1, xyz0, pts0o, B, N0, 32, N1, xyz1, pts1, ws, stream);

  TBp tr1 = mkTB(36);
  run_tb(tr1, xyz1, pts1, pts1o, B, N1, 64, ws, stream);

  SAl sa2[2] = {{F(13), F(14), F(16), F(15), 67,  128},
                {F(17), F(18), F(20), F(19), 128, 128}};
  run_sa(sa2, xyz1, pts1o, B, N1, 64, N2, xyz2, pts2, ws, stream);

  TBp tr2 = mkTB(51);
  run_tb(tr2, xyz2, pts2, (float*)d_out, B, N2, 128, ws, stream);
}
